// CompleteCascadePredictionModel_13297218748850
// MI455X (gfx1250) — matrix operand order repaired and checked by reading, not compiled, not run on hardware
//
#include <hip/hip_runtime.h>

// ---------------- problem constants (from reference) ----------------
#define BB    4
#define NN    5000
#define TT    16
#define FF    64
#define EE    80000
#define HHD   4            // attention heads
#define CCH   16           // channels per head
#define SSD   64           // stream dim = HHD*CCH
#define BN    (BB*NN)      // 20000 nodes
#define ETOT  (BB*EE + BN) // 340000 edges incl. self loops
#define NS    (BN*SSD)     // 1,280,000 elements per state buffer
#define NEG_SLOPE 0.2f
#define EPS_SM 1e-16f
#define EPS_LN 1e-5f

typedef __attribute__((ext_vector_type(16))) _Float16 v16h;
typedef __attribute__((ext_vector_type(8)))  _Float16 v8h;
typedef __attribute__((ext_vector_type(8)))  float    v8f;

// ---------------- WMMA fragment loaders from f16 row-major mirrors ----------------
// A 16x32 f16 (contiguous k order, as in load_b_h): lane<16 -> row M=lane, halves = K0..7 | K8..15
//                             lane>=16 -> row M=lane-16, halves = K16..23 | K24..31
// With f16 row-major storage this is two contiguous 16-byte loads per lane.
__device__ __forceinline__ v16h load_a_h(const _Float16* __restrict__ base, int ld,
                                         int kk, int lane) {
  const int m  = lane & 15;
  const int hi = (lane >> 4) & 1;
  const _Float16* row = base + m * ld + kk;
  v8h lo = *(const v8h*)(row + (hi ? 16 : 0));
  v8h hv = *(const v8h*)(row + (hi ? 24 : 8));
  return __builtin_shufflevector(lo, hv, 0, 1, 2, 3, 4, 5, 6, 7,
                                 8, 9, 10, 11, 12, 13, 14, 15);
}

// B 32x16 f16 where B[k,n] = W16[(n0+n)*ldw + kk + k] (W row-major [out,in]):
// lane<16 -> col n=lane, K=0..15; lane>=16 -> K=16..31.  One contiguous 32-byte load.
__device__ __forceinline__ v16h load_b_h(const _Float16* __restrict__ W, int ldw,
                                         int kk, int n0, int lane) {
  const int n  = lane & 15;
  const int kb = (lane & 16) ? 16 : 0;
  return *(const v16h*)(W + (n0 + n) * ldw + kk + kb);
}

__device__ __forceinline__ float sigm(float x) { return 1.f / (1.f + __expf(-x)); }

__device__ __forceinline__ void atomicMaxF32(float* addr, float val) {
  unsigned int* ua = (unsigned int*)addr;
  unsigned int old = *ua;
  while (__uint_as_float(old) < val) {
    unsigned int assumed = old;
    old = atomicCAS(ua, assumed, __float_as_uint(val));
    if (old == assumed) break;
  }
}

__device__ __forceinline__ int2 edge_nodes(const int* __restrict__ ei, int e) {
  int s, d;
  if (e < BB * EE) {
    int b = e / EE, j = e - b * EE;
    s = ei[j]      + b * NN;
    d = ei[EE + j] + b * NN;
  } else {
    s = d = e - BB * EE;   // self loop
  }
  return make_int2(s, d);
}

// ---------------- generic f32 -> f16 mirror ----------------
__global__ void k_f32_to_f16(const float* __restrict__ src, _Float16* __restrict__ dst,
                             int n) {
  int i = blockIdx.x * blockDim.x + threadIdx.x;
  if (i < n) dst[i] = (_Float16)src[i];
}

// ---------------- GEMM: xt = x_t @ lin_w^T  (one wave -> 16x64 tile, 8 WMMA) --------
__global__ void k_gemm_xt(const _Float16* __restrict__ A, const _Float16* __restrict__ W,
                          float* __restrict__ Y) {
  const int lane = threadIdx.x & 31;
  const int wave = threadIdx.x >> 5;            // 0..1
  const int m0 = (blockIdx.x * 2 + wave) * 16;  // 625 blocks * 2 waves = 1250 tiles
  __builtin_prefetch(A + (m0 + 16) * 64, 0, 1);
  v8f acc[4] = {};
#pragma unroll
  for (int kk = 0; kk < 64; kk += 32) {
    v16h a = load_a_h(A + m0 * 64, 64, kk, lane);
#pragma unroll
    for (int nt = 0; nt < 4; ++nt) {
      v16h b = load_b_h(W, 64, kk, nt * 16, lane);
      acc[nt] = __builtin_amdgcn_wmma_f32_16x16x32_f16(false, a, false, b, (short)0,
                                                       acc[nt], false, false);
    }
  }
  const int mo = (lane >> 4) * 8;
  const int n  = lane & 15;
#pragma unroll
  for (int nt = 0; nt < 4; ++nt)
#pragma unroll
    for (int r = 0; r < 8; ++r)
      Y[(m0 + mo + r) * 64 + nt * 16 + n] = acc[nt][r];
}

// ------- GEMM: gates = sp@Wih^T + h@Whh^T + bih + bhh (wave -> 16x64, 16 WMMA) -------
__global__ void k_gemm_gates(const _Float16* __restrict__ SP, const _Float16* __restrict__ H,
                             const _Float16* __restrict__ Wih, const _Float16* __restrict__ Whh,
                             const float* __restrict__ bih, const float* __restrict__ bhh,
                             float* __restrict__ Gt) {
  const int lane = threadIdx.x & 31;
  const int wave = threadIdx.x >> 5;   // 0..3 -> 64-wide quadrant of the 256 gates
  const int m0 = blockIdx.x * 16;
  const int nq = wave * 64;
  __builtin_prefetch(SP + (m0 + 16) * 64, 0, 1);
  __builtin_prefetch(H + (m0 + 16) * 64, 0, 1);
  v8f acc[4] = {};
#pragma unroll
  for (int kk = 0; kk < 64; kk += 32) {
    v16h a = load_a_h(SP + m0 * 64, 64, kk, lane);
#pragma unroll
    for (int j = 0; j < 4; ++j) {
      v16h b = load_b_h(Wih, 64, kk, nq + j * 16, lane);
      acc[j] = __builtin_amdgcn_wmma_f32_16x16x32_f16(false, a, false, b, (short)0,
                                                      acc[j], false, false);
    }
  }
#pragma unroll
  for (int kk = 0; kk < 64; kk += 32) {
    v16h a = load_a_h(H + m0 * 64, 64, kk, lane);
#pragma unroll
    for (int j = 0; j < 4; ++j) {
      v16h b = load_b_h(Whh, 64, kk, nq + j * 16, lane);
      acc[j] = __builtin_amdgcn_wmma_f32_16x16x32_f16(false, a, false, b, (short)0,
                                                      acc[j], false, false);
    }
  }
  const int mo = (lane >> 4) * 8;
  const int n  = lane & 15;
#pragma unroll
  for (int j = 0; j < 4; ++j) {
    const int col = nq + j * 16 + n;
    const float bias = bih[col] + bhh[col];
#pragma unroll
    for (int r = 0; r < 8; ++r)
      Gt[(m0 + mo + r) * 256 + col] = acc[j][r] + bias;
  }
}

// --- GEMM: z1 = relu(concat(hf,hm,hs) @ W1^T + b1) (wave -> 16x96, 36 WMMA) ---------
__global__ void k_gemm_z1(const _Float16* __restrict__ hf, const _Float16* __restrict__ hm,
                          const _Float16* __restrict__ hs, const _Float16* __restrict__ W1,
                          const float* __restrict__ b1, float* __restrict__ Z1) {
  const int lane = threadIdx.x & 31;
  const int wave = threadIdx.x >> 5;            // 0..1
  const int m0 = (blockIdx.x * 2 + wave) * 16;
  const _Float16* srcs[3] = { hf, hm, hs };
  v8f acc[6] = {};
#pragma unroll
  for (int s = 0; s < 3; ++s) {
    const _Float16* src = srcs[s];
    __builtin_prefetch(src + (m0 + 16) * 64, 0, 1);
#pragma unroll
    for (int kk = 0; kk < 64; kk += 32) {
      v16h a = load_a_h(src + m0 * 64, 64, kk, lane);
#pragma unroll
      for (int j = 0; j < 6; ++j) {
        v16h b = load_b_h(W1, 192, s * 64 + kk, j * 16, lane);
        acc[j] = __builtin_amdgcn_wmma_f32_16x16x32_f16(false, a, false, b, (short)0,
                                                        acc[j], false, false);
      }
    }
  }
  const int mo = (lane >> 4) * 8;
  const int n  = lane & 15;
#pragma unroll
  for (int j = 0; j < 6; ++j) {
    const float bias = b1[j * 16 + n];
#pragma unroll
    for (int r = 0; r < 8; ++r) {
      float v = acc[j][r] + bias;
      Z1[(m0 + mo + r) * 96 + j * 16 + n] = v > 0.f ? v : 0.f;
    }
  }
}

// ---------------- GAT segment-softmax pipeline (f32 + atomics) ----------------
__global__ void k_gat_init(float* __restrict__ sp, float* __restrict__ amax,
                           float* __restrict__ den, const float* __restrict__ gbias) {
  int i = blockIdx.x * blockDim.x + threadIdx.x;
  if (i < BN * SSD) sp[i] = gbias[i & 63];       // seed with bias -> final add fused
  if (i < BN * HHD) { amax[i] = -1e30f; den[i] = 0.f; }
}

__global__ void k_edge_logits(const float* __restrict__ xt, const int* __restrict__ ei,
                              const float* __restrict__ asrc, const float* __restrict__ adst,
                              float* __restrict__ ae, float* __restrict__ amax) {
  int e = blockIdx.x * blockDim.x + threadIdx.x;
  if (e >= ETOT) return;
  int2 sd = edge_nodes(ei, e);
  const float* xs = xt + sd.x * 64;
  const float* xd = xt + sd.y * 64;
#pragma unroll
  for (int h = 0; h < HHD; ++h) {
    float acc = 0.f;
#pragma unroll
    for (int c = 0; c < CCH; ++c)
      acc += xs[h * 16 + c] * asrc[h * 16 + c] + xd[h * 16 + c] * adst[h * 16 + c];
    acc = acc > 0.f ? acc : NEG_SLOPE * acc;     // leaky relu
    ae[e * 4 + h] = acc;
    atomicMaxF32(&amax[sd.y * 4 + h], acc);
  }
}

__global__ void k_edge_exp(const int* __restrict__ ei, float* __restrict__ ae,
                           const float* __restrict__ amax, float* __restrict__ den) {
  int e = blockIdx.x * blockDim.x + threadIdx.x;
  if (e >= ETOT) return;
  int2 sd = edge_nodes(ei, e);
#pragma unroll
  for (int h = 0; h < HHD; ++h) {
    float ex = __expf(ae[e * 4 + h] - amax[sd.y * 4 + h]);
    ae[e * 4 + h] = ex;
    atomicAdd(&den[sd.y * 4 + h], ex);
  }
}

__global__ void k_edge_agg(const int* __restrict__ ei, const float* __restrict__ xt,
                           const float* __restrict__ ae, const float* __restrict__ den,
                           float* __restrict__ sp) {
  int idx = blockIdx.x * blockDim.x + threadIdx.x;
  if (idx >= ETOT * 64) return;
  int e = idx >> 6, c = idx & 63, h = c >> 4;
  int2 sd = edge_nodes(ei, e);
  float alpha = ae[e * 4 + h] / (den[sd.y * 4 + h] + EPS_SM);
  atomicAdd(&sp[sd.y * 64 + c], xt[sd.x * 64 + c] * alpha);
}

// -------- LSTM elementwise + LayerNorm (wave32 per node); writes f32 + f16 h --------
__global__ void k_lstm_ln(const float* __restrict__ Gt, float* __restrict__ Hst,
                          _Float16* __restrict__ H16, float* __restrict__ Cst,
                          const float* __restrict__ lng, const float* __restrict__ lnb) {
  const int node = blockIdx.x * (blockDim.x >> 5) + (threadIdx.x >> 5);
  const int lane = threadIdx.x & 31;
  if (node >= BN) return;
  const float* g = Gt + node * 256;
  float hn[2];
#pragma unroll
  for (int j = 0; j < 2; ++j) {
    int s = lane + j * 32;
    float iv = sigm(g[s]);
    float fv = sigm(g[64 + s]);
    float gv = tanhf(g[128 + s]);
    float ov = sigm(g[192 + s]);
    float cn = fv * Cst[node * 64 + s] + iv * gv;
    Cst[node * 64 + s] = cn;
    hn[j] = ov * tanhf(cn);
  }
  float sum = hn[0] + hn[1];
#pragma unroll
  for (int off = 16; off > 0; off >>= 1) sum += __shfl_xor(sum, off, 32);
  const float mu = sum * (1.f / 64.f);
  float d0 = hn[0] - mu, d1 = hn[1] - mu;
  float vs = d0 * d0 + d1 * d1;
#pragma unroll
  for (int off = 16; off > 0; off >>= 1) vs += __shfl_xor(vs, off, 32);
  const float inv = rsqrtf(vs * (1.f / 64.f) + EPS_LN);
  float o0 = d0 * inv * lng[lane]      + lnb[lane];
  float o1 = d1 * inv * lng[lane + 32] + lnb[lane + 32];
  Hst[node * 64 + lane]      = o0;
  Hst[node * 64 + lane + 32] = o1;
  H16[node * 64 + lane]      = (_Float16)o0;
  H16[node * 64 + lane + 32] = (_Float16)o1;
}

// ---------------- attention mix: softmax(z1 @ W2^T + b2), blend streams -------------
__global__ void k_attn_w(const float* __restrict__ Z1, const float* __restrict__ W2,
                         const float* __restrict__ b2, float* __restrict__ wmix) {
  int node = blockIdx.x * blockDim.x + threadIdx.x;
  if (node >= BN) return;
  const float* z = Z1 + node * 96;
  float l[3];
#pragma unroll
  for (int j = 0; j < 3; ++j) {
    float acc = b2[j];
    for (int k = 0; k < 96; ++k) acc += z[k] * W2[j * 96 + k];
    l[j] = acc;
  }
  float mx = fmaxf(l[0], fmaxf(l[1], l[2]));
  float e0 = __expf(l[0] - mx), e1 = __expf(l[1] - mx), e2 = __expf(l[2] - mx);
  float s = e0 + e1 + e2;
  wmix[node * 3 + 0] = e0 / s;
  wmix[node * 3 + 1] = e1 / s;
  wmix[node * 3 + 2] = e2 / s;
}

__global__ void k_out(const float* __restrict__ hf, const float* __restrict__ hm,
                      const float* __restrict__ hs, const float* __restrict__ wmix,
                      float* __restrict__ out, int t) {
  int idx = blockIdx.x * blockDim.x + threadIdx.x;
  if (idx >= BN * 64) return;
  int node = idx >> 6, c = idx & 63;
  const float* w = wmix + node * 3;
  // output layout [B,N,T,S] -> ((b*N+n)*T + t)*S + c
  out[(node * TT + t) * 64 + c] = w[0] * hf[idx] + w[1] * hm[idx] + w[2] * hs[idx];
}

// ---------------- host orchestration ----------------
extern "C" void kernel_launch(void* const* d_in, const int* in_sizes, int n_in,
                              void* d_out, int out_size, void* d_ws, size_t ws_size,
                              hipStream_t stream) {
  (void)in_sizes; (void)n_in; (void)out_size; (void)ws_size;
  const float* x_seq    = (const float*)d_in[0];
  const float* lin_w    = (const float*)d_in[1];
  const float* att_src  = (const float*)d_in[2];
  const float* att_dst  = (const float*)d_in[3];
  const float* gat_bias = (const float*)d_in[4];
  const float* w_ih     = (const float*)d_in[5];
  const float* w_hh     = (const float*)d_in[6];
  const float* b_ih     = (const float*)d_in[7];
  const float* b_hh     = (const float*)d_in[8];
  const float* ln_g     = (const float*)d_in[9];
  const float* ln_b     = (const float*)d_in[10];
  const float* attn_w1  = (const float*)d_in[11];
  const float* attn_b1  = (const float*)d_in[12];
  const float* attn_w2  = (const float*)d_in[13];
  const float* attn_b2  = (const float*)d_in[14];
  const int*   ei       = (const int*)d_in[15];
  float* out = (float*)d_out;
  float* ws  = (float*)d_ws;

  // -------- workspace carve-up (all offsets multiples of 64 floats = 256 B) --------
  size_t off = 0;
  float* hst[3];  for (int k = 0; k < 3; ++k) { hst[k] = ws + off; off += NS; }
  float* cst[3];  for (int k = 0; k < 3; ++k) { cst[k] = ws + off; off += NS; }
  _Float16* h16[3];
  for (int k = 0; k < 3; ++k) { h16[k] = (_Float16*)(ws + off); off += NS / 2; }
  float* xtf   = ws + off; off += NS;                 // f32 GAT features (edge kernels)
  float* sp    = ws + off; off += NS;                 // f32 GAT output (atomics)
  float* gates = ws + off; off += (size_t)BN * 256;   // reused for z1/wmix
  float* ae    = ws + off; off += (size_t)ETOT * 4;
  float* amax  = ws + off; off += (size_t)BN * 4;
  float* den   = ws + off; off += (size_t)BN * 4;
  _Float16* sp16  = (_Float16*)(ws + off); off += NS / 2;
  _Float16* xts16 = (_Float16*)(ws + off); off += NS / 2;
  _Float16* lw16  = (_Float16*)(ws + off); off += 3 * 64 * 64 / 2;
  _Float16* wih16 = (_Float16*)(ws + off); off += 3 * 256 * 64 / 2;
  _Float16* whh16 = (_Float16*)(ws + off); off += 3 * 256 * 64 / 2;
  _Float16* w1_16 = (_Float16*)(ws + off); off += 96 * 192 / 2;
  float* z1   = gates;                         // 20000*96 (gates dead by then)
  float* wmix = gates + (size_t)BN * 96;       // 20000*3

  // zero h/c f32 states AND the f16 h mirrors (contiguous region)
  hipMemsetAsync(ws, 0, (size_t)(6 * NS + 3 * (NS / 2)) * sizeof(float), stream);

  // one-time weight f16 mirrors
  k_f32_to_f16<<<(3 * 64 * 64 + 255) / 256, 256, 0, stream>>>(lin_w, lw16, 3 * 64 * 64);
  k_f32_to_f16<<<(3 * 256 * 64 + 255) / 256, 256, 0, stream>>>(w_ih, wih16, 3 * 256 * 64);
  k_f32_to_f16<<<(3 * 256 * 64 + 255) / 256, 256, 0, stream>>>(w_hh, whh16, 3 * 256 * 64);
  k_f32_to_f16<<<(96 * 192 + 255) / 256, 256, 0, stream>>>(attn_w1, w1_16, 96 * 192);

  for (int t = 0; t < TT; ++t) {
    const float* x_t = x_seq + (size_t)t * BN * FF;
    k_f32_to_f16<<<(BN * 64 + 255) / 256, 256, 0, stream>>>(x_t, xts16, BN * 64);
    for (int k = 0; k < 3; ++k) {
      if (k == 1 && (t % 15) != 0) continue;   // medium: t = 0, 15
      if (k == 2 && t != 0) continue;          // slow:   t = 0
      k_gat_init<<<(BN * 64 + 255) / 256, 256, 0, stream>>>(sp, amax, den, gat_bias + k * 64);
      k_gemm_xt<<<BN / 32, 64, 0, stream>>>(xts16, lw16 + k * 64 * 64, xtf);
      k_edge_logits<<<(ETOT + 255) / 256, 256, 0, stream>>>(xtf, ei, att_src + k * 64,
                                                            att_dst + k * 64, ae, amax);
      k_edge_exp<<<(ETOT + 255) / 256, 256, 0, stream>>>(ei, ae, amax, den);
      k_edge_agg<<<(ETOT * 64 + 255) / 256, 256, 0, stream>>>(ei, xtf, ae, den, sp);
      k_f32_to_f16<<<(BN * 64 + 255) / 256, 256, 0, stream>>>(sp, sp16, BN * 64);
      k_gemm_gates<<<BN / 16, 128, 0, stream>>>(sp16, h16[k], wih16 + k * 256 * 64,
                                                whh16 + k * 256 * 64, b_ih + k * 256,
                                                b_hh + k * 256, gates);
      k_lstm_ln<<<BN / 8, 256, 0, stream>>>(gates, hst[k], h16[k], cst[k],
                                            ln_g + k * 64, ln_b + k * 64);
    }
    k_gemm_z1<<<BN / 32, 64, 0, stream>>>(h16[0], h16[1], h16[2], w1_16, attn_b1, z1);
    k_attn_w<<<(BN + 255) / 256, 256, 0, stream>>>(z1, attn_w2, attn_b2, wmix);
    k_out<<<(BN * 64 + 255) / 256, 256, 0, stream>>>(hst[0], hst[1], hst[2], wmix, out, t);
  }
}
